// OsSoluModel_74620761800928
// MI455X (gfx1250) — matrix operand order repaired and checked by reading, not compiled, not run on hardware
//
#include <hip/hip_runtime.h>

// ---------------------------------------------------------------------------
// CDNA5 (gfx1250) SoLU transformer forward, bf16 WMMA everywhere.
// ---------------------------------------------------------------------------

typedef __attribute__((ext_vector_type(16))) __bf16          v16bf;
typedef __attribute__((ext_vector_type(16))) unsigned short  v16u;
typedef __attribute__((ext_vector_type(8)))  float           v8f;

#define DMODEL 1024
#define NHEADS 16
#define HS     64
#define FFDIM  4096
#define SEQ    1024
#define BATCH  2
#define NTOK   (BATCH * SEQ)      // 2048
#define VOCAB  50257
#define NEGV   (-100000.0f)

__device__ __forceinline__ unsigned short f2bf(float f) {
  unsigned int u = __float_as_uint(f);
  u += 0x7FFFu + ((u >> 16) & 1u);          // round-to-nearest-even
  return (unsigned short)(u >> 16);
}

__device__ __forceinline__ v8f wmma_bf16(v16u a, v16u b, v8f c) {
  return __builtin_amdgcn_wmma_f32_16x16x32_bf16(
      false, __builtin_bit_cast(v16bf, a),
      false, __builtin_bit_cast(v16bf, b),
      (short)0, c, false, false);
}

// ---------------------------------------------------------------------------
// h = W_E[tok] + W_pos[s]
// ---------------------------------------------------------------------------
__global__ void embed_kernel(const int* __restrict__ tok,
                             const float* __restrict__ WE,
                             const float* __restrict__ Wpos,
                             float* __restrict__ h) {
  int idx = blockIdx.x * 256 + threadIdx.x;   // [0, NTOK*DMODEL)
  int t = idx >> 10, d = idx & 1023;
  int s = t & (SEQ - 1);
  h[idx] = WE[(size_t)tok[t] * DMODEL + d] + Wpos[(size_t)s * DMODEL + d];
}

// ---------------------------------------------------------------------------
// LayerNorm (eps 1e-5), f32 in -> bf16 out. One 256-thread block per row.
// ---------------------------------------------------------------------------
__global__ void ln_kernel(const float* __restrict__ x,
                          const float* __restrict__ g,
                          const float* __restrict__ bb,
                          unsigned short* __restrict__ out) {
  __shared__ float red[256];
  int row = blockIdx.x, t = threadIdx.x;
  const float* xr = x + (size_t)row * DMODEL;
  float v[4], s = 0.f;
#pragma unroll
  for (int i = 0; i < 4; ++i) { v[i] = xr[t + i * 256]; s += v[i]; }
  red[t] = s; __syncthreads();
  for (int off = 128; off > 0; off >>= 1) {
    if (t < off) red[t] += red[t + off];
    __syncthreads();
  }
  float mu = red[0] * (1.f / DMODEL);
  __syncthreads();
  s = 0.f;
#pragma unroll
  for (int i = 0; i < 4; ++i) { float d = v[i] - mu; s += d * d; }
  red[t] = s; __syncthreads();
  for (int off = 128; off > 0; off >>= 1) {
    if (t < off) red[t] += red[t + off];
    __syncthreads();
  }
  float rstd = rsqrtf(red[0] * (1.f / DMODEL) + 1e-5f);
#pragma unroll
  for (int i = 0; i < 4; ++i) {
    int c = t + i * 256;
    out[(size_t)row * DMODEL + c] = f2bf((v[i] - mu) * rstd * g[c] + bb[c]);
  }
}

// ---------------------------------------------------------------------------
// SoLU: out = m * softmax(m) along FF=4096, f32 in -> bf16 out.
// ---------------------------------------------------------------------------
__global__ void solu_kernel(const float* __restrict__ m,
                            unsigned short* __restrict__ out) {
  __shared__ float red[256];
  int row = blockIdx.x, t = threadIdx.x;
  const float* mr = m + (size_t)row * FFDIM;
  float v[16], mx = -1e30f;
#pragma unroll
  for (int i = 0; i < 16; ++i) { v[i] = mr[t + i * 256]; mx = fmaxf(mx, v[i]); }
  red[t] = mx; __syncthreads();
  for (int off = 128; off > 0; off >>= 1) {
    if (t < off) red[t] = fmaxf(red[t], red[t + off]);
    __syncthreads();
  }
  mx = red[0]; __syncthreads();
  float e[16], s = 0.f;
#pragma unroll
  for (int i = 0; i < 16; ++i) { e[i] = __expf(v[i] - mx); s += e[i]; }
  red[t] = s; __syncthreads();
  for (int off = 128; off > 0; off >>= 1) {
    if (t < off) red[t] += red[t + off];
    __syncthreads();
  }
  float inv = 1.f / red[0];
#pragma unroll
  for (int i = 0; i < 16; ++i) {
    out[(size_t)row * FFDIM + t + i * 256] = f2bf(v[i] * e[i] * inv);
  }
}

// ---------------------------------------------------------------------------
// Pack f32 [token, h*64+d] -> bf16 [b, h, s, d] for attention.
// ---------------------------------------------------------------------------
__global__ void pack_kernel(const float* __restrict__ src,
                            unsigned short* __restrict__ dst) {
  int idx = blockIdx.x * 256 + threadIdx.x;   // [0, NTOK*DMODEL)
  int t = idx >> 10, c = idx & 1023;
  int b = t >> 10, s = t & 1023;
  int hh = c >> 6, d = c & 63;
  dst[(((size_t)(b * NHEADS + hh)) * SEQ + s) * HS + d] = f2bf(src[idx]);
}

// ---------------------------------------------------------------------------
// Flash attention, one wave (32 threads) per (b, head, 16-row q-tile).
// Mask applied before /sqrt(64), matching the reference.
// ---------------------------------------------------------------------------
__global__ __launch_bounds__(32)
void attn_kernel(const unsigned short* __restrict__ q,
                 const unsigned short* __restrict__ k,
                 const unsigned short* __restrict__ v,
                 unsigned short* __restrict__ o) {
  __shared__ unsigned short kls[32][64];
  __shared__ unsigned short vls[32][64];
  __shared__ unsigned short pls[16][32];

  int blk  = blockIdx.x;            // 0 .. B*NH*(S/16)-1
  int qt   = blk & 63;              // q tile index
  int bh   = blk >> 6;              // b*16 + head
  int b    = bh >> 4, head = bh & 15;
  int lane = threadIdx.x, hi = lane >> 4, ln = lane & 15;

  const unsigned short* qbase = q + ((size_t)bh * SEQ + qt * 16) * HS;
  const unsigned short* kbase = k + (size_t)bh * SEQ * HS;
  const unsigned short* vbase = v + (size_t)bh * SEQ * HS;

  // Q fragments (A-layout): lane row m=ln, K mapping hi*16+e
  v16u qa[2];
#pragma unroll
  for (int kk = 0; kk < 2; ++kk)
#pragma unroll
    for (int e = 0; e < 16; ++e) {
      int d = kk * 32 + hi * 16 + e;
      qa[kk][e] = qbase[ln * HS + d];
    }

  float mrow[8], lrow[8];
  v8f acc[4];
#pragma unroll
  for (int r = 0; r < 8; ++r) { mrow[r] = -1e30f; lrow[r] = 0.f; }
#pragma unroll
  for (int nt = 0; nt < 4; ++nt)
#pragma unroll
    for (int r = 0; r < 8; ++r) acc[nt][r] = 0.f;

  int npairs = qt / 2 + 1;          // pairs of 16-key tiles covering 0..qt
  for (int p = 0; p < npairs; ++p) {
    int k0 = p * 32;
    // cooperatively stage 32 K rows and 32 V rows (one row per lane)
#pragma unroll 8
    for (int i = 0; i < 64; ++i) {
      kls[lane][i] = kbase[(size_t)(k0 + lane) * HS + i];
      vls[lane][i] = vbase[(size_t)(k0 + lane) * HS + i];
    }
    __syncthreads();

    // scores: two 16x16 tiles, K-dim = 64 = 2 WMMA steps each
    v8f sc[2];
#pragma unroll
    for (int tl = 0; tl < 2; ++tl) {
#pragma unroll
      for (int r = 0; r < 8; ++r) sc[tl][r] = 0.f;
#pragma unroll
      for (int kk = 0; kk < 2; ++kk) {
        v16u bfr;
#pragma unroll
        for (int e = 0; e < 16; ++e)
          bfr[e] = kls[tl * 16 + ln][kk * 32 + hi * 16 + e];
        sc[tl] = wmma_bf16(qa[kk], bfr, sc[tl]);
      }
    }

    // online softmax update (rows hi*8+r, cols across 16 lanes of this half)
    float p0[8], p1[8], scl[8];
#pragma unroll
    for (int r = 0; r < 8; ++r) {
      int qg = qt * 16 + hi * 8 + r;
      float s0 = (qg < k0 + ln)      ? NEGV : sc[0][r];
      float s1 = (qg < k0 + 16 + ln) ? NEGV : sc[1][r];
      s0 *= 0.125f; s1 *= 0.125f;
      float rm = fmaxf(s0, s1);
      for (int off = 8; off >= 1; off >>= 1) rm = fmaxf(rm, __shfl_xor(rm, off, 32));
      float nm = fmaxf(mrow[r], rm);
      scl[r] = __expf(mrow[r] - nm);
      mrow[r] = nm;
      p0[r] = __expf(s0 - nm);
      p1[r] = __expf(s1 - nm);
      float rs = p0[r] + p1[r];
      for (int off = 8; off >= 1; off >>= 1) rs += __shfl_xor(rs, off, 32);
      lrow[r] = lrow[r] * scl[r] + rs;
    }
#pragma unroll
    for (int r = 0; r < 8; ++r) {
      int rw = hi * 8 + r;
      pls[rw][ln]      = f2bf(p0[r]);
      pls[rw][16 + ln] = f2bf(p1[r]);
    }
#pragma unroll
    for (int nt = 0; nt < 4; ++nt)
#pragma unroll
      for (int r = 0; r < 8; ++r) acc[nt][r] *= scl[r];
    __syncthreads();

    // O += P(16x32) x V(32x64)
    v16u pa;
#pragma unroll
    for (int e = 0; e < 16; ++e)
      pa[e] = pls[ln][hi * 16 + e];
#pragma unroll
    for (int nt = 0; nt < 4; ++nt) {
      v16u vb;
#pragma unroll
      for (int e = 0; e < 16; ++e) vb[e] = vls[hi * 16 + e][nt * 16 + ln];
      acc[nt] = wmma_bf16(pa, vb, acc[nt]);
    }
    __syncthreads();
  }

  // write o[b, s, h*64+d] bf16
#pragma unroll
  for (int nt = 0; nt < 4; ++nt)
#pragma unroll
    for (int r = 0; r < 8; ++r) {
      size_t tokrow = (size_t)b * SEQ + qt * 16 + hi * 8 + r;
      o[tokrow * DMODEL + head * HS + nt * 16 + ln] =
          f2bf(acc[nt][r] / lrow[r]);
    }
}

// ---------------------------------------------------------------------------
// Tiled WMMA GEMM: C[M,N] = A[M,K](bf16) x W(f32 -> bf16) + bias + resid
//   bt==0: W is [K,N] row-major;  bt==1: W is [N,K] row-major (B^T, W_E)
// Block: 128 threads = 4 waves; 64(M) x 128(N) tile; K step 32.
// Each wave owns a 32x64 sub-tile: 8 WMMAs per K-step.
// Weight tile stored transposed in LDS (sBt[n][k], 80B rows) so B-fragment
// reads are two 16B-aligned ds_load_b128 each, bank-conflict free.
// Grid: x = M tiles (fast) so M-blocks sharing a weight tile co-reside -> L2 reuse.
// ---------------------------------------------------------------------------
__global__ __launch_bounds__(128)
void gemm_kernel(const unsigned short* __restrict__ A,
                 const float* __restrict__ W,
                 const float* __restrict__ bias,
                 const float* __restrict__ resid,
                 float* __restrict__ C,
                 int M, int N, int K, int bt) {
  __shared__ unsigned short sA[64][40];     // 80B rows: 16B aligned chunks
  __shared__ unsigned short sBt[128][40];   // transposed weight tile

  int t  = threadIdx.x;
  int m0 = blockIdx.x * 64, n0 = blockIdx.y * 128;
  int wid = t >> 5, lane = t & 31, hi = lane >> 4, ln = lane & 15;
  int wm = (wid >> 1) * 32;                 // wave M offset (0/32)
  int wn = (wid & 1) * 64;                  // wave N offset (0/64)

  v8f acc[2][4];
#pragma unroll
  for (int i = 0; i < 2; ++i)
#pragma unroll
    for (int j = 0; j < 4; ++j)
#pragma unroll
      for (int r = 0; r < 8; ++r) acc[i][j][r] = 0.f;

  for (int k0 = 0; k0 < K; k0 += 32) {
    {   // stage A tile 64x32 bf16: 2 threads per row, 16 elems each
      int row = t >> 1, half = t & 1;
      const unsigned short* src = A + (size_t)(m0 + row) * K + k0 + half * 16;
#pragma unroll
      for (int i = 0; i < 16; ++i) sA[row][half * 16 + i] = src[i];
    }
    if (!bt) {   // W is [K,N]: 4 threads per k-row, 32 contiguous n each
      int row = t >> 2;                 // k within tile: 0..31
      int c0  = (t & 3) * 32;           // n group
      const float* src = W + (size_t)(k0 + row) * N + n0 + c0;
#pragma unroll
      for (int i = 0; i < 32; ++i) {
        int gn = n0 + c0 + i;
        sBt[c0 + i][row] = (gn < N) ? f2bf(src[i]) : (unsigned short)0;
      }
    } else {     // W is [N,K]: one n-row per thread, 32 contiguous k
      int nr = t;                       // 0..127
      int gn = n0 + nr;
      if (gn < N) {
        const float* src = W + (size_t)gn * K + k0;
#pragma unroll
        for (int i = 0; i < 32; ++i) sBt[nr][i] = f2bf(src[i]);
      } else {
#pragma unroll
        for (int i = 0; i < 32; ++i) sBt[nr][i] = 0;
      }
    }
    __syncthreads();

    v16u af[2];
#pragma unroll
    for (int s = 0; s < 2; ++s)
#pragma unroll
      for (int e = 0; e < 16; ++e)
        af[s][e] = sA[wm + s * 16 + ln][hi * 16 + e];

#pragma unroll
    for (int j = 0; j < 4; ++j) {
      v16u bf;
#pragma unroll
      for (int e = 0; e < 16; ++e)
        bf[e] = sBt[wn + j * 16 + ln][hi * 16 + e];
#pragma unroll
      for (int i = 0; i < 2; ++i)
        acc[i][j] = wmma_bf16(af[i], bf, acc[i][j]);
    }
    __syncthreads();
  }

#pragma unroll
  for (int i = 0; i < 2; ++i)
#pragma unroll
    for (int j = 0; j < 4; ++j)
#pragma unroll
      for (int r = 0; r < 8; ++r) {
        int gm = m0 + wm + i * 16 + hi * 8 + r;
        int gn = n0 + wn + j * 16 + ln;
        if (gn < N) {
          float val = acc[i][j][r];
          if (bias)  val += bias[gn];
          if (resid) val += resid[(size_t)gm * N + gn];
          C[(size_t)gm * N + gn] = val;
        }
      }
}

// ---------------------------------------------------------------------------
// Host driver
// ---------------------------------------------------------------------------
extern "C" void kernel_launch(void* const* d_in, const int* in_sizes, int n_in,
                              void* d_out, int out_size, void* d_ws, size_t ws_size,
                              hipStream_t stream) {
  (void)in_sizes; (void)n_in; (void)out_size; (void)ws_size;
  const int*   tokens = (const int*)  d_in[0];
  const float* W_E    = (const float*)d_in[1];
  const float* W_pos  = (const float*)d_in[2];
  const float* ln1_g  = (const float*)d_in[3];
  const float* ln1_b  = (const float*)d_in[4];
  const float* Wq     = (const float*)d_in[5];
  const float* bq     = (const float*)d_in[6];
  const float* Wk     = (const float*)d_in[7];
  const float* bk     = (const float*)d_in[8];
  const float* Wv     = (const float*)d_in[9];
  const float* bv     = (const float*)d_in[10];
  const float* Wo     = (const float*)d_in[11];
  const float* bo     = (const float*)d_in[12];
  const float* ln2_g  = (const float*)d_in[13];
  const float* ln2_b  = (const float*)d_in[14];
  const float* Wi     = (const float*)d_in[15];
  const float* bi     = (const float*)d_in[16];
  const float* Wo2    = (const float*)d_in[17];
  const float* bo2    = (const float*)d_in[18];
  const float* lnf_g  = (const float*)d_in[19];
  const float* lnf_b  = (const float*)d_in[20];
  float* out = (float*)d_out;

  char* p = (char*)d_ws;
  float*          h  = (float*)p;          p += (size_t)NTOK * DMODEL * 4;
  unsigned short* xn = (unsigned short*)p; p += (size_t)NTOK * DMODEL * 2;
  float*          qf = (float*)p;          p += (size_t)NTOK * DMODEL * 4;
  float*          kf = (float*)p;          p += (size_t)NTOK * DMODEL * 4;
  float*          vf = (float*)p;          p += (size_t)NTOK * DMODEL * 4;
  unsigned short* qb = (unsigned short*)p; p += (size_t)NTOK * DMODEL * 2;
  unsigned short* kb = (unsigned short*)p; p += (size_t)NTOK * DMODEL * 2;
  unsigned short* vb = (unsigned short*)p; p += (size_t)NTOK * DMODEL * 2;
  unsigned short* ob = (unsigned short*)p; p += (size_t)NTOK * DMODEL * 2;
  float*          mf = (float*)p;          p += (size_t)NTOK * FFDIM * 4;
  unsigned short* mb = (unsigned short*)p; p += (size_t)NTOK * FFDIM * 2;

  const int elem_blocks = NTOK * DMODEL / 256;      // 8192
  embed_kernel<<<elem_blocks, 256, 0, stream>>>(tokens, W_E, W_pos, h);

  dim3 gg(NTOK / 64, DMODEL / 128);                 // (32, 8)
  dim3 gi(NTOK / 64, FFDIM / 128);                  // (32, 32)

  for (int l = 0; l < 2; ++l) {
    size_t wdd = (size_t)l * DMODEL * DMODEL;
    ln_kernel<<<NTOK, 256, 0, stream>>>(h, ln1_g + l * DMODEL, ln1_b + l * DMODEL, xn);

    gemm_kernel<<<gg, 128, 0, stream>>>(xn, Wq + wdd, bq + l * DMODEL, nullptr, qf,
                                        NTOK, DMODEL, DMODEL, 0);
    gemm_kernel<<<gg, 128, 0, stream>>>(xn, Wk + wdd, bk + l * DMODEL, nullptr, kf,
                                        NTOK, DMODEL, DMODEL, 0);
    gemm_kernel<<<gg, 128, 0, stream>>>(xn, Wv + wdd, bv + l * DMODEL, nullptr, vf,
                                        NTOK, DMODEL, DMODEL, 0);

    pack_kernel<<<elem_blocks, 256, 0, stream>>>(qf, qb);
    pack_kernel<<<elem_blocks, 256, 0, stream>>>(kf, kb);
    pack_kernel<<<elem_blocks, 256, 0, stream>>>(vf, vb);

    attn_kernel<<<BATCH * NHEADS * (SEQ / 16), 32, 0, stream>>>(qb, kb, vb, ob);

    gemm_kernel<<<gg, 128, 0, stream>>>(ob, Wo + wdd, bo + l * DMODEL, h, h,
                                        NTOK, DMODEL, DMODEL, 0);

    ln_kernel<<<NTOK, 256, 0, stream>>>(h, ln2_g + l * DMODEL, ln2_b + l * DMODEL, xn);

    gemm_kernel<<<gi, 128, 0, stream>>>(xn, Wi + (size_t)l * DMODEL * FFDIM,
                                        bi + l * FFDIM, nullptr, mf,
                                        NTOK, FFDIM, DMODEL, 0);

    solu_kernel<<<NTOK, 256, 0, stream>>>(mf, mb);

    gemm_kernel<<<gg, 128, 0, stream>>>(mb, Wo2 + (size_t)l * FFDIM * DMODEL,
                                        bo2 + l * DMODEL, h, h,
                                        NTOK, DMODEL, FFDIM, 0);
  }

  ln_kernel<<<NTOK, 256, 0, stream>>>(h, lnf_g, lnf_b, xn);

  dim3 gl(NTOK / 64, (VOCAB + 127) / 128);          // (32, 393)
  gemm_kernel<<<gl, 128, 0, stream>>>(xn, W_E, nullptr, nullptr, out,
                                      NTOK, VOCAB, DMODEL, 1);
}
